// unidirLSTM_Layer_52939766891147
// MI455X (gfx1250) — matrix operand order repaired and checked by reading, not compiled, not run on hardware
//
#include <hip/hip_runtime.h>

// ---------------------------------------------------------------------------
// Types for CDNA5 WMMA (wave32): bf16 A/B fragments, f32 accumulators.
// ---------------------------------------------------------------------------
typedef __attribute__((ext_vector_type(16))) __bf16 v16bf;
typedef __attribute__((ext_vector_type(8)))  __bf16 v8bf;
typedef __attribute__((ext_vector_type(4)))  __bf16 v4bf;
typedef __attribute__((ext_vector_type(8)))  float  v8f;

constexpr int B_   = 32;
constexpr int T_   = 2048;
constexpr int H_   = 256;        // input features (= K of proj GEMM)
constexpr int HID_ = 256;        // hidden size
constexpr int G4_  = 4 * HID_;   // 1024 gate columns (i,f,g,o)
constexpr int ROWS = B_ * T_;    // 65536 rows of the input-projection GEMM

constexpr int NWG = 16;          // persistent workgroups for the scan
constexpr int CW  = HID_ / NWG;  // 16 hidden channels owned per workgroup

__device__ __forceinline__ v16bf join8(v8bf lo, v8bf hi) {
  return __builtin_shufflevector(lo, hi, 0, 1, 2, 3, 4, 5, 6, 7,
                                         8, 9, 10, 11, 12, 13, 14, 15);
}

__device__ __forceinline__ float fast_sigmoid(float x) {
  return __builtin_amdgcn_rcpf(1.0f + __expf(-x));
}
__device__ __forceinline__ float fast_tanh(float x) {
  // tanh(x) = 2*sigmoid(2x) - 1
  return __builtin_fmaf(2.0f, __builtin_amdgcn_rcpf(1.0f + __expf(-2.0f * x)),
                        -1.0f);
}

// Generic pointer to LDS byte offset (flat LDS aperture keeps the offset in
// the low 32 bits; hardware uses addr[31:0] as LDS_ADDR).
__device__ __forceinline__ unsigned lds_off(const void* p) {
  return (unsigned)(uintptr_t)p;
}

// ---------------------------------------------------------------------------
// Prep kernels: fp32 -> bf16 once, so the GEMM can stage raw bytes with
// async global->LDS copies (no VALU conversion in the hot loop).
// ---------------------------------------------------------------------------
__global__ __launch_bounds__(256) void cvt_x(const float* __restrict__ X,
                                             __bf16* __restrict__ Xb) {
  size_t i = ((size_t)blockIdx.x * 256 + threadIdx.x) * 4;
  const float4 v = *(const float4*)&X[i];
  v4bf b;
  b[0] = (__bf16)v.x; b[1] = (__bf16)v.y; b[2] = (__bf16)v.z; b[3] = (__bf16)v.w;
  *(v4bf*)&Xb[i] = b;
}

// WiT[n][k] = (bf16) Wi[k][n]   (1 MB total: cost is noise)
__global__ __launch_bounds__(256) void cvt_wiT(const float* __restrict__ Wi,
                                               __bf16* __restrict__ WiT) {
  int id = blockIdx.x * 256 + threadIdx.x;   // 0 .. 256*1024-1
  int k = id & (H_ - 1);
  int n = id >> 8;
  WiT[(size_t)n * H_ + k] = (__bf16)Wi[(size_t)k * G4_ + n];
}

__global__ void init_h(const float* __restrict__ h0, __bf16* __restrict__ hbuf) {
  int i = blockIdx.x * 256 + threadIdx.x;
  if (i < B_ * HID_) hbuf[i] = (__bf16)h0[i];
}

// ---------------------------------------------------------------------------
// Kernel 1: x_proj[65536,1024] = Xb[65536,256] @ WiT^T, stored bf16.
// 128x128 block tile, 8 waves, double-buffered LDS tiles filled with
// GLOBAL_LOAD_ASYNC_TO_LDS_B128 (ASYNCcnt), 8 WMMAs per wave per K-step.
// ---------------------------------------------------------------------------
#define BM 128
#define BN 128
#define BK 32
#define LDA 40   // BK + 8 pad: 80B row stride (16B aligned, bank-shifted)

__global__ __launch_bounds__(256) void xproj_gemm(
    const __bf16* __restrict__ Xb, const __bf16* __restrict__ WiT,
    __bf16* __restrict__ XP) {
  __shared__ __bf16 As[2][BM][LDA];   // As[buf][m][k]
  __shared__ __bf16 Bs[2][BN][LDA];   // Bs[buf][n][k] (K contiguous)

  const int tid  = threadIdx.x;
  const int lane = tid & 31;
  const int wv   = tid >> 5;
  const int wm   = wv & 1;   // M half (0..1)
  const int wn   = wv >> 1;  // N quarter (0..3)

  const int mbase = blockIdx.y * BM;
  const int nbase = blockIdx.x * BN;

  // Per-thread async-copy chunk: 2 rows x one 16B chunk for A and B each.
  const int cr = tid >> 2;                       // row 0..63 (then +64)
  const int cc = tid & 3;                        // 16B chunk within 64B row
  const unsigned aB   = lds_off(&As[0][0][0]);
  const unsigned bB   = lds_off(&Bs[0][0][0]);
  const unsigned bufS = (unsigned)(BM * LDA * sizeof(__bf16));  // 10240
  const unsigned ro   = (unsigned)(cr * (LDA * 2) + cc * 16);
  const unsigned ro2  = ro + (unsigned)(64 * LDA * 2);

  const __bf16* gA  = Xb  + (size_t)(mbase + cr) * H_ + cc * 8;
  const __bf16* gA2 = gA  + (size_t)64 * H_;
  const __bf16* gB  = WiT + (size_t)(nbase + cr) * H_ + cc * 8;
  const __bf16* gB2 = gB  + (size_t)64 * H_;

  auto issue = [&](int k0, int buf) {
    unsigned d = buf * bufS;
    uint64_t a0 = (uint64_t)(gA + k0),  a1 = (uint64_t)(gA2 + k0);
    uint64_t b0 = (uint64_t)(gB + k0),  b1 = (uint64_t)(gB2 + k0);
    unsigned la = aB + d + ro, la2 = aB + d + ro2;
    unsigned lb = bB + d + ro, lb2 = bB + d + ro2;
    asm volatile("global_load_async_to_lds_b128 %0, %1, off"
                 :: "v"(la),  "v"(a0) : "memory");
    asm volatile("global_load_async_to_lds_b128 %0, %1, off"
                 :: "v"(la2), "v"(a1) : "memory");
    asm volatile("global_load_async_to_lds_b128 %0, %1, off"
                 :: "v"(lb),  "v"(b0) : "memory");
    asm volatile("global_load_async_to_lds_b128 %0, %1, off"
                 :: "v"(lb2), "v"(b1) : "memory");
  };

  v8f acc[4][2] = {};
  const int mrow  = lane & 15;
  const int koffA = (lane & 16) ? 8 : 0;   // A: lanes>=16 hold K {8..15,24..31}
  const int koffB = (lane & 16) ? 8 : 0;   // B: lanes>=16 hold K {8..15,24..31}

  issue(0, 0);
  for (int kk = 0; kk < 8; ++kk) {
    const int buf = kk & 1;
    if (kk < 7) {
      issue((kk + 1) * BK, buf ^ 1);               // prefetch next tile
      asm volatile("s_wait_asynccnt 4" ::: "memory");  // current tile landed
    } else {
      asm volatile("s_wait_asynccnt 0" ::: "memory");
    }
    __syncthreads();

    v16bf af[4];
    for (int mt = 0; mt < 4; ++mt) {
      const __bf16* pa = &As[buf][wm * 64 + mt * 16 + mrow][koffA];
      af[mt] = join8(*(const v8bf*)pa, *(const v8bf*)(pa + 16));
    }
    v16bf bfr[2];
    for (int nt = 0; nt < 2; ++nt) {
      const __bf16* pb = &Bs[buf][wn * 32 + nt * 16 + mrow][koffB];
      bfr[nt] = join8(*(const v8bf*)pb, *(const v8bf*)(pb + 16));
    }
    for (int mt = 0; mt < 4; ++mt)
      for (int nt = 0; nt < 2; ++nt)
        acc[mt][nt] = __builtin_amdgcn_wmma_f32_16x16x32_bf16(
            false, af[mt], false, bfr[nt], (short)0, acc[mt][nt], false, false);
    __syncthreads();   // tile consumed; its buffer may be overwritten next
  }

  // ---- epilogue: C/D layout -> global bf16 ----
  const int rb = (lane & 16) ? 8 : 0;
  for (int mt = 0; mt < 4; ++mt)
    for (int nt = 0; nt < 2; ++nt) {
      int col  = nbase + wn * 32 + nt * 16 + (lane & 15);
      int row0 = mbase + wm * 64 + mt * 16 + rb;
      for (int r = 0; r < 8; ++r)
        XP[(size_t)(row0 + r) * G4_ + col] = (__bf16)acc[mt][nt][r];
    }
}

// ---------------------------------------------------------------------------
// Kernel 2: persistent recurrent scan. 16 WGs x 256 threads. WG `wg` owns
// hidden channels [wg*16, wg*16+16) and their 4 gate columns (64 cols).
// Wh slice lives in LDS for all 2048 steps; cell state c lives in registers.
// Per step: z[32x64] = h[32x256] @ WhS  (8 waves x 8 WMMAs), gates in VALU,
// h_new broadcast via global bf16 double buffer + per-step atomic barrier.
// ---------------------------------------------------------------------------
__global__ __launch_bounds__(256) void lstm_scan(
    const float* __restrict__ Wh, const float* __restrict__ bias,
    const float* __restrict__ c0, const __bf16* __restrict__ XP,
    __bf16* __restrict__ hbuf, int* __restrict__ syncc,
    float* __restrict__ out) {
  __shared__ __bf16 WhS[64][264];  // [local col][k], col-major, 33.8 KB
  __shared__ float  zS[32][64];    // gate pre-activations staging
  __shared__ float  bS[64];

  const int tid  = threadIdx.x;
  const int lane = tid & 31;
  const int wv   = tid >> 5;
  const int c0ch = blockIdx.x * CW;

  // ---- one-time: stage Wh slice (bf16, col-major) + bias slice ----
  for (int i = tid; i < 64 * 256; i += 256) {
    int lc   = i & 63;                                   // local col: g*16+j
    int k    = i >> 6;
    int gcol = (lc >> 4) * HID_ + c0ch + (lc & 15);      // gate-major layout
    WhS[lc][k] = (__bf16)Wh[(size_t)k * G4_ + gcol];
  }
  if (tid < 64) bS[tid] = bias[(tid >> 4) * HID_ + c0ch + (tid & 15)];

  // cell state: 32 batches x 16 channels = 512 values, 2 per thread
  float creg[2];
  for (int u = 0; u < 2; ++u) {
    int idx = tid + u * 256;
    creg[u] = c0[(idx >> 4) * HID_ + c0ch + (idx & 15)];
  }
  __syncthreads();

  const int mt    = wv & 1;                 // 2 M tiles (batch 0-15 / 16-31)
  const int nt    = wv >> 1;                // 4 N tiles (16 gate cols each)
  const int mrow  = mt * 16 + (lane & 15);
  const int koffA = (lane & 16) ? 8 : 0;
  const int koffB = (lane & 16) ? 8 : 0;
  const __bf16* brow = &WhS[nt * 16 + (lane & 15)][0];
  const int zcol  = nt * 16 + (lane & 15);
  const int zrow0 = mt * 16 + ((lane & 16) ? 8 : 0);

  for (int t = 0; t < T_; ++t) {
    if (t > 0) {
      if (tid == 0) {
        while (__hip_atomic_load(&syncc[t], __ATOMIC_ACQUIRE,
                                 __HIP_MEMORY_SCOPE_AGENT) < NWG)
          __builtin_amdgcn_s_sleep(1);
      }
      __syncthreads();
    }

    // ---- z tile: h @ WhS.  A fragments straight from global bf16 h buffer
    //      (row-major [batch][256]: K-contiguous -> two b128 loads each). ----
    const __bf16* hrow = hbuf + (size_t)(t & 1) * (B_ * HID_) + mrow * HID_;
    v8f acc = {};
    for (int ks = 0; ks < 8; ++ks) {
      const __bf16* pa = hrow + ks * 32 + koffA;
      v16bf a  = join8(*(const v8bf*)pa, *(const v8bf*)(pa + 16));
      const __bf16* pb = brow + ks * 32 + koffB;
      v16bf bm = join8(*(const v8bf*)pb, *(const v8bf*)(pb + 16));
      acc = __builtin_amdgcn_wmma_f32_16x16x32_bf16(false, a, false, bm,
                                                    (short)0, acc, false,
                                                    false);
    }
    for (int r = 0; r < 8; ++r) zS[zrow0 + r][zcol] = acc[r];
    __syncthreads();

    // ---- elementwise gates: 512 (batch,channel) pairs, 2 per thread ----
    for (int u = 0; u < 2; ++u) {
      int idx = tid + u * 256;
      int bb  = idx >> 4;
      int j   = idx & 15;
      size_t xb = ((size_t)bb * T_ + t) * G4_ + c0ch + j;
      float zi = zS[bb][j]      + (float)XP[xb]            + bS[j];
      float zf = zS[bb][16 + j] + (float)XP[xb + HID_]     + bS[16 + j];
      float zg = zS[bb][32 + j] + (float)XP[xb + 2 * HID_] + bS[32 + j];
      float zo = zS[bb][48 + j] + (float)XP[xb + 3 * HID_] + bS[48 + j];
      float ig = fast_sigmoid(zi);
      float fg = fast_sigmoid(zf);
      float gg = fast_tanh(zg);
      float og = fast_sigmoid(zo);
      float cn = __builtin_fmaf(fg, creg[u], ig * gg);
      float hn = og * fast_tanh(cn);
      creg[u] = cn;
      // broadcast h_{t+1} (bf16) + emit output y[:,t,:] (f32)
      hbuf[(size_t)((t + 1) & 1) * (B_ * HID_) + bb * HID_ + c0ch + j] =
          (__bf16)hn;
      out[(size_t)2 * B_ * HID_ + ((size_t)bb * T_ + t) * HID_ + c0ch + j] = hn;
      if (t == T_ - 1) {
        out[bb * HID_ + c0ch + j]             = hn;  // hT
        out[B_ * HID_ + bb * HID_ + c0ch + j] = cn;  // cT
      }
      if (t + 1 < T_) __builtin_prefetch(&XP[xb + G4_], 0, 1);
    }
    __threadfence();
    __syncthreads();
    if (tid == 0)
      __hip_atomic_fetch_add(&syncc[t + 1], 1, __ATOMIC_RELEASE,
                             __HIP_MEMORY_SCOPE_AGENT);
  }
}

// ---------------------------------------------------------------------------
// Launch. Workspace layout:
//   [0,16KB)              per-step barrier counters
//   [16KB,48KB)           bf16 h double buffer
//   [64KB, +128MB)        bf16 x_proj
//   [+128MB, +32MB)       bf16 X
//   [+..., +512KB)        bf16 Wi^T
// ---------------------------------------------------------------------------
extern "C" void kernel_launch(void* const* d_in, const int* in_sizes, int n_in,
                              void* d_out, int out_size, void* d_ws,
                              size_t ws_size, hipStream_t stream) {
  (void)in_sizes; (void)n_in; (void)out_size; (void)ws_size;
  const float* X  = (const float*)d_in[0];
  const float* Wi = (const float*)d_in[1];
  const float* Wh = (const float*)d_in[2];
  const float* bv = (const float*)d_in[3];
  const float* h0 = (const float*)d_in[4];
  const float* c0 = (const float*)d_in[5];
  float* out = (float*)d_out;

  char*   ws    = (char*)d_ws;
  int*    syncc = (int*)ws;
  __bf16* hbuf  = (__bf16*)(ws + 16384);
  __bf16* XP    = (__bf16*)(ws + 65536);
  __bf16* Xb    = (__bf16*)(ws + 65536 + (size_t)ROWS * G4_ * 2);
  __bf16* WiT   = (__bf16*)((char*)Xb + (size_t)ROWS * H_ * 2);

  hipMemsetAsync(syncc, 0, 16384, stream);
  cvt_x<<<dim3(ROWS * H_ / 1024), dim3(256), 0, stream>>>(X, Xb);
  cvt_wiT<<<dim3(H_ * G4_ / 256), dim3(256), 0, stream>>>(Wi, WiT);
  init_h<<<dim3((B_ * HID_ + 255) / 256), dim3(256), 0, stream>>>(h0, hbuf);
  xproj_gemm<<<dim3(G4_ / BN, ROWS / BM), dim3(256), 0, stream>>>(Xb, WiT, XP);
  lstm_scan<<<dim3(NWG), dim3(256), 0, stream>>>(Wh, bv, c0, XP, hbuf, syncc,
                                                 out);
}
